// FlowVQAutoencoder_6038724018961
// MI455X (gfx1250) — matrix operand order repaired and checked by reading, not compiled, not run on hardware
//
#include <hip/hip_runtime.h>
#include <hip/hip_bf16.h>

// ---------------- WMMA types (CDNA5 gfx1250, wave32) ----------------
typedef __attribute__((ext_vector_type(16))) __bf16 v16bf;
typedef __attribute__((ext_vector_type(8)))  float  v8f;

union ABu { v16bf v; unsigned short u[16]; unsigned int d[8]; };

__device__ __forceinline__ unsigned short f2bf(float f) {
    union { float f; unsigned int u; } c; c.f = f;
    unsigned int u = c.u;
    unsigned int r = (u + 0x7FFFu + ((u >> 16) & 1u)) >> 16;  // RNE fp32->bf16
    return (unsigned short)r;
}
// pack two fp32 -> one dword of two bf16 (lo, hi), RNE
__device__ __forceinline__ unsigned int pack2bf(float flo, float fhi) {
    union { float f; unsigned int u; } a, b; a.f = flo; b.f = fhi;
    unsigned int lo = (a.u + 0x7FFFu + ((a.u >> 16) & 1u)) >> 16;
    unsigned int hi = (b.u + 0x7FFFu + ((b.u >> 16) & 1u)) & 0xFFFF0000u;
    return hi | lo;
}

// ---------------- Fully-static conv / convT implicit GEMM ----------------
// out[m, co] = sum_k A[m,k] * W[k, co] (+bias, optional BN, LeakyReLU(0.2))
// MODE 0 conv:  iy = oy*S + ky - PAD
// MODE 1 convT: ty = oy + PAD - ky; valid iff ty>=0 && ty%S==0 && ty/S<IH
// Fast path: whole tap's weights (CI x 16*NT bf16, <=32KB) staged in LDS with
// +16B column padding (bank-conflict-free b128 reads); 2 barriers per TAP.
template<int N, int IH, int IW, int CI, int OH, int OW, int CO,
         int KH, int KW, int S, int PAD, int MODE, int ACT, int NT>
__global__ __launch_bounds__(256) void conv_wmma(
        const float* __restrict__ in, const float* __restrict__ w,
        const float* __restrict__ bia, const float* __restrict__ gam,
        const float* __restrict__ off, float* __restrict__ out) {
    constexpr int M = N * OH * OW;
    constexpr int K = KH * KW * CI;
    constexpr int NCOLS = 16 * NT;
    constexpr bool FAST = (CI & 31) == 0;
    constexpr int CIP = FAST ? (CI + 8) : 40;   // padded col pitch (ushorts)
    static_assert(M % 128 == 0, "block covers exactly 8 full 16-row tiles");
    static_assert((CIP * 2) % 16 == 0, "b128-aligned LDS columns");

    __shared__ unsigned short Bs[NCOLS * CIP];

    const int lane = threadIdx.x & 31;
    const int wave = threadIdx.x >> 5;
    const int half = lane >> 4;
    const int ln16 = lane & 15;

    const int m0  = (blockIdx.x * 8 + wave) * 16;
    const int co0 = blockIdx.y * NCOLS;

    const int mr  = m0 + ln16;
    const int n   = mr / (OH * OW);
    const int rem = mr - n * (OH * OW);
    const int oy  = rem / OW;                // OW power of two -> shift
    const int ox  = rem - oy * OW;

    v8f acc[NT] = {};

    if constexpr (FAST) {
        #pragma unroll 1
        for (int tap = 0; tap < KH * KW; ++tap) {
            const int ky = tap / KW;
            const int kx = tap - ky * KW;
            int iy, ix; bool ok;
            if constexpr (MODE == 0) {
                iy = oy * S + ky - PAD;
                ix = ox * S + kx - PAD;
                ok = (iy >= 0) & (iy < IH) & (ix >= 0) & (ix < IW);
            } else {
                int ty = oy + PAD - ky, tx = ox + PAD - kx;
                ok = (ty >= 0) & (tx >= 0) && ((ty % S) == 0) && ((tx % S) == 0);
                iy = ty / S; ix = tx / S;
                ok = ok & (iy < IH) & (ix < IW);
            }
            const float* src = ok ? &in[(((long)n * IH + iy) * IW + ix) * CI] : nullptr;

            // ---- stage the whole tap's weights: read coalesced over cols ----
            #pragma unroll
            for (int i = threadIdx.x; i < CI * NCOLS; i += 256) {
                int ci = i / NCOLS, col = i - ci * NCOLS;   // NCOLS pow2 -> shifts
                float wv;
                if constexpr (CO % 16 == 0) wv = w[(long)(tap * CI + ci) * CO + (co0 + col)];
                else wv = (co0 + col < CO) ? w[(long)(tap * CI + ci) * CO + (co0 + col)] : 0.f;
                Bs[col * CIP + ci] = f2bf(wv);
            }
            __syncthreads();
            if (threadIdx.x == 0 && tap + 1 < KH * KW)
                __builtin_prefetch(&w[(long)(tap + 1) * CI * CO + co0], 0, 1);

            #pragma unroll 1
            for (int cib = 0; cib < CI; cib += 32) {
                // ---- A fragment: two contiguous 8-float runs per lane ----
                ABu a;
                if (ok) {
                    const float4* p4 = (const float4*)(src + cib);
                    float4 r0 = p4[half * 2 + 0];
                    float4 r1 = p4[half * 2 + 1];
                    float4 r2 = p4[half * 2 + 4];
                    float4 r3 = p4[half * 2 + 5];
                    a.d[0] = pack2bf(r0.x, r0.y); a.d[1] = pack2bf(r0.z, r0.w);
                    a.d[2] = pack2bf(r1.x, r1.y); a.d[3] = pack2bf(r1.z, r1.w);
                    a.d[4] = pack2bf(r2.x, r2.y); a.d[5] = pack2bf(r2.z, r2.w);
                    a.d[6] = pack2bf(r3.x, r3.y); a.d[7] = pack2bf(r3.z, r3.w);
                } else {
                    #pragma unroll
                    for (int e = 0; e < 8; ++e) a.d[e] = 0;
                }
                // ---- NT unrolled WMMAs reuse one A fragment ----
                #pragma unroll
                for (int t = 0; t < NT; ++t) {
                    ABu b;
                    const int base = (t * 16 + ln16) * CIP + cib + half * 8;
                    #pragma unroll
                    for (int e = 0; e < 16; ++e) b.u[e] = Bs[base + (e < 8 ? e : e + 8)];
                    acc[t] = __builtin_amdgcn_wmma_f32_16x16x32_bf16(
                                 false, a.v, false, b.v, (short)0, acc[t], false, false);
                }
            }
            __syncthreads();
        }
    } else {
        // ============ GENERIC PATH (enc1: CI=2, K=50) ============
        #pragma unroll 1
        for (int k0 = 0; k0 < K; k0 += 32) {
            #pragma unroll
            for (int i = threadIdx.x; i < NCOLS * 32; i += 256) {
                int col = i >> 5, k = i & 31;
                int gk = k0 + k, co = co0 + col;
                float wv = (gk < K && co < CO) ? w[(long)gk * CO + co] : 0.f;
                Bs[col * CIP + k] = f2bf(wv);
            }
            __syncthreads();
            ABu a;
            #pragma unroll
            for (int e = 0; e < 16; ++e) {
                int k = k0 + half * 8 + (e < 8 ? e : e + 8);
                float v = 0.f;
                if (k < K) {
                    int ky = k / (KW * CI);
                    int r2 = k - ky * (KW * CI);
                    int kx = r2 / CI;
                    int ci = r2 - kx * CI;
                    int iy = oy * S + ky - PAD;
                    int ix = ox * S + kx - PAD;
                    if ((iy >= 0) & (iy < IH) & (ix >= 0) & (ix < IW))
                        v = in[(((long)n * IH + iy) * IW + ix) * CI + ci];
                }
                a.u[e] = f2bf(v);
            }
            #pragma unroll
            for (int t = 0; t < NT; ++t) {
                ABu b;
                const int base = (t * 16 + ln16) * CIP + half * 8;
                #pragma unroll
                for (int e = 0; e < 16; ++e) b.u[e] = Bs[base + (e < 8 ? e : e + 8)];
                acc[t] = __builtin_amdgcn_wmma_f32_16x16x32_bf16(
                             false, a.v, false, b.v, (short)0, acc[t], false, false);
            }
            __syncthreads();
        }
    }

    // ---- epilogue: bias, folded BN, LeakyReLU; C/D layout m = r + half*8 ----
    #pragma unroll
    for (int t = 0; t < NT; ++t) {
        const int co = co0 + t * 16 + ln16;
        bool cok = true;
        if constexpr (CO % 16 != 0) cok = (co < CO);
        if (cok) {
            const float bias = bia[co];
            float gg = 1.f, oo = 0.f;
            if constexpr (ACT) { gg = gam[co]; oo = off[co]; }
            #pragma unroll
            for (int r = 0; r < 8; ++r) {
                int m = m0 + half * 8 + r;
                float v = acc[t][r] + bias;
                if constexpr (ACT) { v = v * gg + oo; v = v > 0.f ? v : 0.2f * v; }
                out[(long)m * CO + co] = v;
            }
        }
    }
}

// ---------------- VQ: codebook column norms ----------------
__global__ void vq_cnorm(const float* __restrict__ cb, float* __restrict__ cn) {
    int k = blockIdx.x * blockDim.x + threadIdx.x;
    if (k < 512) {
        float s = 0.f;
        #pragma unroll 8
        for (int d = 0; d < 64; ++d) { float v = cb[d * 512 + k]; s += v * v; }
        cn[k] = s;
    }
}

// ---------------- VQ: distance GEMM (WMMA, codebook in LDS) + argmin ----------------
__global__ __launch_bounds__(256) void vq_argmin(const float* __restrict__ z,
                                                 const float* __restrict__ cb,
                                                 const float* __restrict__ cn,
                                                 int* __restrict__ idx) {
    __shared__ unsigned short CbL[512 * 64];  // 64 KB [col][d] bf16
    __shared__ float norms[8][16];

    const int lane = threadIdx.x & 31;
    const int wave = threadIdx.x >> 5;
    const int half = lane >> 4;
    const int ln16 = lane & 15;
    const int m0 = (blockIdx.x * 8 + wave) * 16;

    for (int i = threadIdx.x; i < 512 * 64; i += 256) {
        int col = i >> 6, d = i & 63;
        CbL[i] = f2bf(cb[(long)d * 512 + col]);
    }
    if (lane < 16) {
        int m = m0 + lane; float s = 0.f;
        #pragma unroll 8
        for (int d = 0; d < 64; ++d) { float v = z[(long)m * 64 + d]; s += v * v; }
        norms[wave][lane] = s;
    }
    __syncthreads();
    float myn[8];
    #pragma unroll
    for (int r = 0; r < 8; ++r) myn[r] = norms[wave][half * 8 + r];

    const int mr = m0 + ln16;
    ABu alo, ahi;
    {
        const float4* p4 = (const float4*)&z[(long)mr * 64];
        float4 r0 = p4[half*2+0], r1 = p4[half*2+1], r2 = p4[half*2+4], r3 = p4[half*2+5];
        alo.d[0] = pack2bf(r0.x, r0.y); alo.d[1] = pack2bf(r0.z, r0.w);
        alo.d[2] = pack2bf(r1.x, r1.y); alo.d[3] = pack2bf(r1.z, r1.w);
        alo.d[4] = pack2bf(r2.x, r2.y); alo.d[5] = pack2bf(r2.z, r2.w);
        alo.d[6] = pack2bf(r3.x, r3.y); alo.d[7] = pack2bf(r3.z, r3.w);
        float4 h0 = p4[8+half*2+0], h1 = p4[8+half*2+1], h2 = p4[8+half*2+4], h3 = p4[8+half*2+5];
        ahi.d[0] = pack2bf(h0.x, h0.y); ahi.d[1] = pack2bf(h0.z, h0.w);
        ahi.d[2] = pack2bf(h1.x, h1.y); ahi.d[3] = pack2bf(h1.z, h1.w);
        ahi.d[4] = pack2bf(h2.x, h2.y); ahi.d[5] = pack2bf(h2.z, h2.w);
        ahi.d[6] = pack2bf(h3.x, h3.y); ahi.d[7] = pack2bf(h3.z, h3.w);
    }

    float bd[8]; int bi[8];
    #pragma unroll
    for (int r = 0; r < 8; ++r) { bd[r] = 3.4e38f; bi[r] = 0; }

    #pragma unroll 1
    for (int c0 = 0; c0 < 512; c0 += 16) {
        const int col = c0 + ln16;
        const int cbase = col * 64 + half * 8;
        ABu blo, bhi;
        #pragma unroll
        for (int e = 0; e < 16; ++e) { int ke = (e < 8 ? e : e + 8); blo.u[e] = CbL[cbase + ke]; bhi.u[e] = CbL[cbase + 32 + ke]; }
        v8f acc = {};
        acc = __builtin_amdgcn_wmma_f32_16x16x32_bf16(false, alo.v, false, blo.v, (short)0, acc, false, false);
        acc = __builtin_amdgcn_wmma_f32_16x16x32_bf16(false, ahi.v, false, bhi.v, (short)0, acc, false, false);
        const float cno = cn[col];
        #pragma unroll
        for (int r = 0; r < 8; ++r) {
            float dist = myn[r] - 2.f * acc[r] + cno;
            if (dist < bd[r]) { bd[r] = dist; bi[r] = col; }
        }
    }
    #pragma unroll
    for (int r = 0; r < 8; ++r) {
        float d = bd[r]; int i = bi[r];
        #pragma unroll
        for (int off = 8; off; off >>= 1) {
            float od = __shfl_xor(d, off, 16);
            int   oi = __shfl_xor(i, off, 16);
            if (od < d || (od == d && oi < i)) { d = od; i = oi; }
        }
        if (ln16 == 0) { int m = m0 + half * 8 + r; idx[m] = i; }
    }
}

// ---------------- VQ: gather q, loss accumulation, idx out ----------------
__global__ __launch_bounds__(256) void vq_gather_loss(const float* __restrict__ z,
                                                      const float* __restrict__ cb,
                                                      const int* __restrict__ idx,
                                                      float* __restrict__ q,
                                                      float* __restrict__ lacc,
                                                      float* __restrict__ out_idx_f) {
    __shared__ float red[256];
    long t = (long)blockIdx.x * blockDim.x + threadIdx.x;  // exactly M*64 threads
    int m = (int)(t >> 6), d = (int)(t & 63);
    float qv = cb[(long)d * 512 + idx[m]];
    q[t] = qv;
    float dv = qv - z[t];
    float diff2 = dv * dv;
    if (d == 0) out_idx_f[m] = (float)idx[m];
    red[threadIdx.x] = diff2;
    __syncthreads();
    for (int s = 128; s; s >>= 1) {
        if (threadIdx.x < s) red[threadIdx.x] += red[threadIdx.x + s];
        __syncthreads();
    }
    if (threadIdx.x == 0) atomicAdd(lacc, red[0]);
}

__global__ void vq_final(const float* __restrict__ lacc, float* __restrict__ out_loss, float scale) {
    if (threadIdx.x == 0 && blockIdx.x == 0) *out_loss = 1.25f * (*lacc) * scale;
}

// ---------------- host-side layer driver ----------------
template<int N, int IH, int IW, int CI, int OH, int OW, int CO,
         int KH, int KW, int S, int PAD, int MODE, int ACT, int NT>
static void launch_conv(const float* in, const float* w, const float* b,
                        const float* g, const float* o, float* out, hipStream_t stream) {
    constexpr int M = N * OH * OW;
    dim3 grid((M / 16) / 8, (CO + 16 * NT - 1) / (16 * NT));
    conv_wmma<N, IH, IW, CI, OH, OW, CO, KH, KW, S, PAD, MODE, ACT, NT>
        <<<grid, 256, 0, stream>>>(in, w, b, g, o, out);
}

extern "C" void kernel_launch(void* const* d_in, const int* in_sizes, int n_in,
                              void* d_out, int out_size, void* d_ws, size_t ws_size,
                              hipStream_t stream) {
    (void)in_sizes; (void)n_in; (void)out_size; (void)ws_size;
    const float* x   = (const float*)d_in[0];
    const float* ew1 = (const float*)d_in[1];  const float* eb1 = (const float*)d_in[2];
    const float* g1  = (const float*)d_in[3];  const float* o1  = (const float*)d_in[4];
    const float* ew2 = (const float*)d_in[5];  const float* eb2 = (const float*)d_in[6];
    const float* g2  = (const float*)d_in[7];  const float* o2  = (const float*)d_in[8];
    const float* ew3 = (const float*)d_in[9];  const float* eb3 = (const float*)d_in[10];
    const float* g3  = (const float*)d_in[11]; const float* o3  = (const float*)d_in[12];
    const float* ew4 = (const float*)d_in[13]; const float* eb4 = (const float*)d_in[14];
    const float* g4  = (const float*)d_in[15]; const float* o4  = (const float*)d_in[16];
    const float* ew5 = (const float*)d_in[17]; const float* eb5 = (const float*)d_in[18];
    const float* cbk = (const float*)d_in[19];
    const float* dw1 = (const float*)d_in[20]; const float* db1 = (const float*)d_in[21];
    const float* dg1 = (const float*)d_in[22]; const float* do1 = (const float*)d_in[23];
    const float* dw2 = (const float*)d_in[24]; const float* db2 = (const float*)d_in[25];
    const float* dg2 = (const float*)d_in[26]; const float* do2 = (const float*)d_in[27];
    const float* dw3 = (const float*)d_in[28]; const float* db3 = (const float*)d_in[29];
    const float* dg3 = (const float*)d_in[30]; const float* do3 = (const float*)d_in[31];
    const float* dw4 = (const float*)d_in[32]; const float* db4 = (const float*)d_in[33];
    const float* dg4 = (const float*)d_in[34]; const float* do4 = (const float*)d_in[35];
    const float* dw5 = (const float*)d_in[36]; const float* db5 = (const float*)d_in[37];

    // ---- workspace layout (ping-pong activation buffers) ----
    char* wsb = (char*)d_ws;
    size_t off = 0;
    auto take = [&](size_t bytes) { void* p = wsb + off; off += (bytes + 255) & ~(size_t)255; return p; };
    float* bufA = (float*)take((size_t)4 * 384 * 512 * 64 * 4);   // 201 MB
    float* bufB = (float*)take((size_t)4 * 192 * 256 * 128 * 4);  // 100 MB
    float* qbuf = (float*)take((size_t)12288 * 64 * 4);
    float* cn   = (float*)take(512 * 4);
    int*   idx  = (int*)  take(12288 * 4);
    float* lacc = (float*)take(4);

    constexpr int M_VQ = 12288;  // 4*48*64

    // ---- encoder ----  <N, IH, IW, CI, OH, OW, CO, KH, KW, S, PAD, MODE, ACT, NT>
    launch_conv<4, 384, 512,   2, 192, 256,  64, 5, 5, 2, 1, 0, 1, 4>(x,    ew1, eb1, g1, o1, bufA, stream);
    launch_conv<4, 192, 256,  64,  96, 128, 128, 5, 5, 2, 1, 0, 1, 4>(bufA, ew2, eb2, g2, o2, bufB, stream);
    launch_conv<4,  96, 128, 128,  48,  64, 256, 3, 3, 2, 0, 0, 1, 4>(bufB, ew3, eb3, g3, o3, bufA, stream);
    launch_conv<4,  48,  64, 256,  48,  64, 256, 3, 3, 1, 1, 0, 1, 4>(bufA, ew4, eb4, g4, o4, bufB, stream);
    launch_conv<4,  48,  64, 256,  48,  64,  64, 1, 1, 1, 0, 0, 0, 4>(bufB, ew5, eb5, nullptr, nullptr, bufA, stream);
    // bufA = z [4,48,64,64]

    // ---- vector quantization ----
    vq_cnorm<<<2, 256, 0, stream>>>(cbk, cn);
    vq_argmin<<<(M_VQ / 16) / 8, 256, 0, stream>>>(bufA, cbk, cn, idx);
    hipMemsetAsync(lacc, 0, 4, stream);
    float* out_loss  = (float*)d_out + 1572864;
    float* out_idx_f = (float*)d_out + 1572865;
    vq_gather_loss<<<(M_VQ * 64) / 256, 256, 0, stream>>>(
        bufA, cbk, idx, qbuf, lacc, out_idx_f);
    vq_final<<<1, 32, 0, stream>>>(lacc, out_loss, 1.f / (float)(M_VQ * 64));
    // qbuf = z_q (straight-through forward value)

    // ---- decoder ----
    launch_conv<4,  48,  64,  64,  48,  64, 256, 3, 3, 1, 1, 1, 1, 4>(qbuf, dw1, db1, dg1, do1, bufB, stream);
    launch_conv<4,  48,  64, 256,  96, 128, 256, 3, 3, 2, 0, 1, 1, 4>(bufB, dw2, db2, dg2, do2, bufA, stream);
    launch_conv<4,  96, 128, 256, 192, 256, 128, 5, 5, 2, 1, 1, 1, 4>(bufA, dw3, db3, dg3, do3, bufB, stream);
    launch_conv<4, 192, 256, 128, 384, 512,  64, 5, 5, 2, 1, 1, 1, 4>(bufB, dw4, db4, dg4, do4, bufA, stream);
    launch_conv<4, 384, 512,  64, 384, 512,   2, 3, 3, 1, 1, 1, 0, 1>(bufA, dw5, db5, nullptr, nullptr, (float*)d_out, stream);
}
